// InteractAttention_68736656605344
// MI455X (gfx1250) — compile-verified, matrix operand order checked by reading, not run on hardware
//
#include <hip/hip_runtime.h>
#include <hip/hip_bf16.h>

// ---------------------------------------------------------------------------
// Fused InteractAttention for MI455X (gfx1250, wave32, WMMA f16->f32, TDM)
//
// Kernel 1 (pre-pass): convert 6 weight matrices fp32 [K][256] -> f16
//   transposed [256][K] (fragment-major, K-contiguous) into d_ws (~4.3 MB).
// Kernel 2 (fused): 8 waves = 4 M-waves x 2 N-waves per workgroup, 64 rows.
//   B tiles are DMA'd into LDS by the Tensor Data Mover one K-step ahead
//   (double-buffered, TENSORcnt-fenced); A tiles are cvt-staged into LDS.
//   24 v_wmma_f32_16x16x32_f16 per wave per K-step, fragment loads
//   software-pipelined one n-tile ahead via sched_group_barrier.
// ---------------------------------------------------------------------------

#define B_ROWS   32768
#define DIM_C    768
#define DIM_P    2048
#define DIM_H    256
#define MT       64            // rows per workgroup
#define KT       32            // K-step
#define A_STRIDE 40            // halfs per A row (32 + 8 pad -> 80B, 16B aligned)
#define B_ROWH   40            // halfs per B row after TDM padding (32 + 4dw pad)
#define B_MATH   (DIM_H * B_ROWH)      // 10240 halfs per matrix tile
#define B_BUFH   (3 * B_MATH)          // 30720 halfs per double-buffer half

// transposed-f16 weight offsets inside d_ws (units: halfs)
#define OFF_LW1   0u
#define OFF_AW12  196608u
#define OFF_AW2   393216u
#define OFF_LW2   589824u
#define OFF_AW1   1114112u
#define OFF_AW23  1638400u

typedef _Float16 v16h __attribute__((ext_vector_type(16)));
typedef _Float16 v8h  __attribute__((ext_vector_type(8)));
typedef float    v8f  __attribute__((ext_vector_type(8)));
typedef unsigned int u32x4 __attribute__((ext_vector_type(4)));
typedef int          i32x8 __attribute__((ext_vector_type(8)));
typedef int          i32x4 __attribute__((ext_vector_type(4)));

// shared memory carve (bytes)
#define A_OFF    0                            // 2 x 64x40 halfs = 10240
#define B_OFF    10240                        // 2 x 61440 = 122880
#define CB_OFF   (B_OFF + 2 * B_BUFH * 2)     // 133120
#define CW1_OFF  (CB_OFF + MT * DIM_H * 4)    // 198656
#define ST_OFF   (CW1_OFF + MT * DIM_H * 4)   // 264192
#define SMEM_BYTES (ST_OFF + 5 * MT * 4)      // 265472

// sched_group_barrier masks
#define SG_WMMA   0x008
#define SG_DSREAD 0x100

__device__ __forceinline__ float sigf(float x) {
    return 1.0f / (1.0f + __expf(-x));
}

// Fragment per ISA 7.12.2 (16-bit, 16x32 / 32x16): lane&15 = M (or N),
// lane>>4 picks kbase 0/8; halfs 0..7 = K[kbase..+7], 8..15 = K[16+kbase..+7].
__device__ __forceinline__ v16h load_frag(const _Float16* p) {
    union { v16h v; v8h h[2]; } u;
    u.h[0] = *(const v8h*)p;
    u.h[1] = *(const v8h*)(p + 16);
    return u.v;
}

__device__ __forceinline__ float hred16(float v) {
    v += __shfl_xor(v, 1, 32);
    v += __shfl_xor(v, 2, 32);
    v += __shfl_xor(v, 4, 32);
    v += __shfl_xor(v, 8, 32);
    return v;
}

__device__ __forceinline__ v8f wmma16(v16h a, v16h b, v8f c) {
    return __builtin_amdgcn_wmma_f32_16x16x32_f16(false, a, false, b,
                                                  (short)0, c, false, false);
}

// ---------------------------------------------------------------------------
// Tensor Data Mover: DMA a [256 rows x 32 halfs] f16 tile (row stride K halfs)
// from global into LDS at lds_off, padding each 16-dword row with 4 dwords
// (-> 80B LDS row stride, bank-conflict-free fragment loads).
// D# per CDNA5 ISA; 2D tile -> groups 2/3 zero.
// 6-arg builtin form (clang-23 / therock-10.0 headers).
// ---------------------------------------------------------------------------
__device__ __forceinline__ void tdm_load_tile(const _Float16* gsrc,
                                              unsigned lds_off, int K)
{
    const unsigned long long ga = (unsigned long long)(const void*)gsrc;
    u32x4 g0;
    g0[0] = 1u;                                            // count=1 (valid)
    g0[1] = lds_off;                                       // lds_addr (bytes)
    g0[2] = (unsigned)(ga & 0xFFFFFFFFu);                  // global_addr lo
    g0[3] = (unsigned)((ga >> 32) & 0x1FFFFFFu) | (2u << 30);  // hi | type=2
    i32x8 g1;
    g1[0] = (int)((1u << 16)      // data_size = 2 bytes
                | (1u << 20)      // pad_enable
                | (3u << 22)      // pad_interval: every 16 dwords
                | (3u << 25));    // pad_amount: 4 dwords
    g1[1] = (int)(((unsigned)K & 0xFFFFu) << 16);          // tensor_dim0 lo
    g1[2] = (int)(((unsigned)K >> 16) | (256u << 16));     // dim0 hi | dim1 lo
    g1[3] = (int)(32u << 16);                              // dim1 hi | tile_dim0=32
    g1[4] = (int)256;                                      // tile_dim1=256
    g1[5] = (int)K;                                        // dim0_stride lo
    g1[6] = 0;                                             // stride hi | dim1_stride
    g1[7] = 0;
    const i32x4 z4 = {0, 0, 0, 0};
    const i32x8 z8 = {0, 0, 0, 0, 0, 0, 0, 0};
    __builtin_amdgcn_tensor_load_to_lds(g0, g1, z4, z4, z8, 0);
}

// Three GEMMs sharing one A operand. B: TDM double-buffered LDS tiles.
__device__ __forceinline__ void gemm3(
    const float* __restrict__ X, int K,
    const _Float16* __restrict__ W0, const _Float16* __restrict__ W1,
    const _Float16* __restrict__ W2,
    int r0, char* smem, v8f acc[3][8],
    int t, int lane, int wave, int wm, int wn)
{
    _Float16* A_sh = (_Float16*)(smem + A_OFF);
    _Float16* B_sh = (_Float16*)(smem + B_OFF);
    const _Float16* Ws[3] = { W0, W1, W2 };

    const int kbase   = (lane >> 4) * 8;
    const int arow_off = (wm * 16 + (lane & 15)) * A_STRIDE;
    const int lin = t * 8;
    const int arow = lin >> 5;            // staging: row 0..63
    const int akk  = lin & 31;            // staging: k 0..31 (mult of 8)
    const int nsteps = K / KT;

    // ---- prologue: stage step 0 into buffer 0 ----
    {
        const float* src = X + (size_t)(r0 + arow) * K + akk;
        float4 f0 = *(const float4*)src;
        float4 f1 = *(const float4*)(src + 4);
        v8h h;
        h[0] = (_Float16)f0.x; h[1] = (_Float16)f0.y;
        h[2] = (_Float16)f0.z; h[3] = (_Float16)f0.w;
        h[4] = (_Float16)f1.x; h[5] = (_Float16)f1.y;
        h[6] = (_Float16)f1.z; h[7] = (_Float16)f1.w;
        *(v8h*)(A_sh + arow * A_STRIDE + akk) = h;
    }
    if (wave == 0) {
        #pragma unroll
        for (int m = 0; m < 3; ++m)
            tdm_load_tile(Ws[m], (unsigned)(B_OFF + m * B_MATH * 2), K);
    }

    for (int s = 0; s < nsteps; ++s) {
        const int buf = s & 1;
        if (wave == 0) __builtin_amdgcn_s_wait_tensorcnt(0);
        __syncthreads();   // buffers for step s are ready; step s-1 fully read

        if (s + 1 < nsteps) {
            const int k1 = (s + 1) * KT;
            // stage next A slice into the other buffer
            const float* src = X + (size_t)(r0 + arow) * K + k1 + akk;
            float4 f0 = *(const float4*)src;
            float4 f1 = *(const float4*)(src + 4);
            v8h h;
            h[0] = (_Float16)f0.x; h[1] = (_Float16)f0.y;
            h[2] = (_Float16)f0.z; h[3] = (_Float16)f0.w;
            h[4] = (_Float16)f1.x; h[5] = (_Float16)f1.y;
            h[6] = (_Float16)f1.z; h[7] = (_Float16)f1.w;
            *(v8h*)(A_sh + (buf ^ 1) * (MT * A_STRIDE) + arow * A_STRIDE + akk) = h;
            if (s + 2 < nsteps)
                __builtin_prefetch(src + KT, 0, 0);
            // DMA next B slice into the other buffer (wave-level op)
            if (wave == 0) {
                #pragma unroll
                for (int m = 0; m < 3; ++m)
                    tdm_load_tile(Ws[m] + k1,
                        (unsigned)(B_OFF + ((buf ^ 1) * B_BUFH + m * B_MATH) * 2), K);
            }
        }

        // ---- compute on current buffers (1-deep fragment pipeline) ----
        const v16h a = load_frag(A_sh + buf * (MT * A_STRIDE) + arow_off + kbase);
        const _Float16* bb = B_sh + buf * B_BUFH + kbase
                           + (size_t)(wn * 128 + (lane & 15)) * B_ROWH;
        v16h bc0 = load_frag(bb);
        v16h bc1 = load_frag(bb + B_MATH);
        v16h bc2 = load_frag(bb + 2 * B_MATH);
        #pragma unroll
        for (int nt = 0; nt < 8; ++nt) {
            v16h bn0, bn1, bn2;
            if (nt < 7) {
                const int o = (nt + 1) * 16 * B_ROWH;
                bn0 = load_frag(bb + o);
                bn1 = load_frag(bb + o + B_MATH);
                bn2 = load_frag(bb + o + 2 * B_MATH);
            }
            acc[0][nt] = wmma16(a, bc0, acc[0][nt]);
            acc[1][nt] = wmma16(a, bc1, acc[1][nt]);
            acc[2][nt] = wmma16(a, bc2, acc[2][nt]);
            // pin the interleave: 6 DS reads (next n-tile) ahead of 3 WMMAs
            __builtin_amdgcn_sched_group_barrier(SG_DSREAD, 6, 0);
            __builtin_amdgcn_sched_group_barrier(SG_WMMA,   3, 0);
            if (nt < 7) { bc0 = bn0; bc1 = bn1; bc2 = bn2; }
        }
    }
}

// ---------------- pre-pass: fp32 [K][256] -> f16 transposed [256][K] -------
__global__ __launch_bounds__(256) void convert_weights_f16t(
    const float* __restrict__ s0, const float* __restrict__ s1,
    const float* __restrict__ s2, const float* __restrict__ s3,
    const float* __restrict__ s4, const float* __restrict__ s5,
    _Float16* __restrict__ ws)
{
    const float* src; _Float16* dst; int K;
    switch (blockIdx.y) {
        case 0:  src = s0; dst = ws + OFF_LW1;  K = DIM_C; break;
        case 1:  src = s1; dst = ws + OFF_AW12; K = DIM_C; break;
        case 2:  src = s2; dst = ws + OFF_AW2;  K = DIM_C; break;
        case 3:  src = s3; dst = ws + OFF_LW2;  K = DIM_P; break;
        case 4:  src = s4; dst = ws + OFF_AW1;  K = DIM_P; break;
        default: src = s5; dst = ws + OFF_AW23; K = DIM_P; break;
    }
    const int gid = blockIdx.x * 256 + threadIdx.x;
    if (gid < K * DIM_H) {
        const int n = gid & (DIM_H - 1);
        const int k = gid >> 8;
        dst[(size_t)n * K + k] = (_Float16)src[gid];
    }
}

// ---------------- fused main kernel ----------------------------------------
__global__ __launch_bounds__(256) void interact_attention_fused(
    const float* __restrict__ xc,  const float* __restrict__ xp,
    const float* __restrict__ LB1, const float* __restrict__ LB2,
    const _Float16* __restrict__ ws, float* __restrict__ out)
{
    extern __shared__ char smem[];
    float* cb_sh  = (float*)(smem + CB_OFF);    // c_bak  [64][256]
    float* cw1_sh = (float*)(smem + CW1_OFF);   // c_w1   [64][256]
    float* st     = (float*)(smem + ST_OFF);    // row stats 5 x 64
    float* st_sumcw = st;
    float* st_ncb   = st + MT;
    float* st_pw1   = st + 2 * MT;
    float* st_dot   = st + 3 * MT;
    float* st_npb   = st + 4 * MT;

    const int t    = threadIdx.x;
    const int lane = t & 31;
    const int wave = t >> 5;
    const int wm   = wave >> 1;      // 0..3  (M)
    const int wn   = wave & 1;       // 0..1  (N)
    const int r0   = blockIdx.x * MT;
    const int half = lane >> 4;      // 0/1 -> row offset +0/+8 in C tiles

    for (int idx = t; idx < 5 * MT; idx += 256) st[idx] = 0.0f;

    v8f acc[3][8];
    #pragma unroll
    for (int m = 0; m < 3; ++m)
        #pragma unroll
        for (int nt = 0; nt < 8; ++nt)
            acc[m][nt] = (v8f){0.f,0.f,0.f,0.f,0.f,0.f,0.f,0.f};

    // ===================== Phase C: x_content GEMMs ========================
    gemm3(xc, DIM_C, ws + OFF_LW1, ws + OFF_AW12, ws + OFF_AW2,
          r0, smem, acc, t, lane, wave, wm, wn);

    float lcw[8], lncb[8];
    #pragma unroll
    for (int i = 0; i < 8; ++i) { lcw[i] = 0.f; lncb[i] = 0.f; }

    #pragma unroll
    for (int nt = 0; nt < 8; ++nt) {
        const int col = wn * 128 + nt * 16 + (lane & 15);
        const float lb = LB1[col];
        #pragma unroll
        for (int i = 0; i < 8; ++i) {
            const int rl = wm * 16 + i + (half << 3);
            const float cb  = acc[0][nt][i] + lb;
            const float cw  = cb * sigf(acc[1][nt][i]);
            const float cw1 = cb * sigf(acc[2][nt][i]);
            cb_sh [rl * DIM_H + col] = cb;
            cw1_sh[rl * DIM_H + col] = cw1;
            lcw[i]  += cw;
            lncb[i] += cb * cb;
        }
    }
    #pragma unroll
    for (int i = 0; i < 8; ++i) {
        const float rcw = hred16(lcw[i]);
        const float rnc = hred16(lncb[i]);
        if ((lane & 15) == 0) {
            const int rl = wm * 16 + i + (half << 3);
            atomicAdd(&st_sumcw[rl], rcw);
            atomicAdd(&st_ncb[rl],   rnc);
        }
    }
    __syncthreads();   // stats + cb/cw1 visible; staging buffers free for reuse

    // ===================== Phase P: x_pic GEMMs ============================
    #pragma unroll
    for (int m = 0; m < 3; ++m)
        #pragma unroll
        for (int nt = 0; nt < 8; ++nt)
            acc[m][nt] = (v8f){0.f,0.f,0.f,0.f,0.f,0.f,0.f,0.f};

    gemm3(xp, DIM_P, ws + OFF_LW2, ws + OFF_AW1, ws + OFF_AW23,
          r0, smem, acc, t, lane, wave, wm, wn);

    const size_t BH = (size_t)B_ROWS * DIM_H;
    float lpw1[8], ldot[8], lnpb[8];
    #pragma unroll
    for (int i = 0; i < 8; ++i) { lpw1[i] = 0.f; ldot[i] = 0.f; lnpb[i] = 0.f; }

    #pragma unroll
    for (int nt = 0; nt < 8; ++nt) {
        const int col = wn * 128 + nt * 16 + (lane & 15);
        const float lb = LB2[col];
        #pragma unroll
        for (int i = 0; i < 8; ++i) {
            const int rl  = wm * 16 + i + (half << 3);
            const size_t row = (size_t)(r0 + rl);
            const float pb  = acc[0][nt][i] + lb;
            const float pw  = pb * sigf(acc[1][nt][i]);
            const float pw1 = pb * sigf(acc[2][nt][i]);
            const float cbv = cb_sh[rl * DIM_H + col];
            lpw1[i] += pw1;
            ldot[i] += cbv * pb;
            lnpb[i] += pb * pb;
            // c_out = sigmoid(c_bak) * tanh(p_w * mean(c_w))
            const float mcw = st_sumcw[rl] * (1.0f / DIM_H);
            out[row * DIM_H + col] = sigf(cbv) * tanhf(pw * mcw);
            acc[0][nt][i] = pb;           // keep p_bak in registers
        }
    }
    #pragma unroll
    for (int i = 0; i < 8; ++i) {
        const float r1 = hred16(lpw1[i]);
        const float r2 = hred16(ldot[i]);
        const float r3 = hred16(lnpb[i]);
        if ((lane & 15) == 0) {
            const int rl = wm * 16 + i + (half << 3);
            atomicAdd(&st_pw1[rl], r1);
            atomicAdd(&st_dot[rl], r2);
            atomicAdd(&st_npb[rl], r3);
        }
    }
    __syncthreads();

    // ===================== Final: p_out ====================================
    #pragma unroll
    for (int nt = 0; nt < 8; ++nt) {
        const int col = wn * 128 + nt * 16 + (lane & 15);
        #pragma unroll
        for (int i = 0; i < 8; ++i) {
            const int rl  = wm * 16 + i + (half << 3);
            const size_t row = (size_t)(r0 + rl);
            const float pb   = acc[0][nt][i];
            const float cw1v = cw1_sh[rl * DIM_H + col];
            const float mpw1 = st_pw1[rl] * (1.0f / DIM_H);
            const float sim  = st_dot[rl] * rsqrtf(st_ncb[rl] * st_npb[rl]);
            out[BH + row * DIM_H + col] = sigf(pb) * tanhf(cw1v * mpw1) * sim;
        }
    }
}

extern "C" void kernel_launch(void* const* d_in, const int* in_sizes, int n_in,
                              void* d_out, int out_size, void* d_ws, size_t ws_size,
                              hipStream_t stream) {
    const float* xc   = (const float*)d_in[0];
    const float* xp   = (const float*)d_in[1];
    const float* LW1  = (const float*)d_in[2];
    const float* LB1  = (const float*)d_in[3];
    const float* LW2  = (const float*)d_in[4];
    const float* LB2  = (const float*)d_in[5];
    const float* AW1  = (const float*)d_in[6];
    const float* AW12 = (const float*)d_in[7];
    const float* AW2  = (const float*)d_in[8];
    const float* AW23 = (const float*)d_in[9];
    float* out = (float*)d_out;
    _Float16* ws = (_Float16*)d_ws;   // needs >= 4,325,376 bytes

    // pre-pass: weights -> f16, transposed fragment-major
    dim3 cgrid((DIM_P * DIM_H) / 256, 6);
    hipLaunchKernelGGL(convert_weights_f16t, cgrid, dim3(256), 0, stream,
                       LW1, AW12, AW2, LW2, AW1, AW23, ws);

    dim3 grid(B_ROWS / MT);   // 512 workgroups
    dim3 block(256);          // 8 waves (wave32)
    hipLaunchKernelGGL(interact_attention_fused, grid, block, SMEM_BYTES, stream,
                       xc, xp, LB1, LB2, ws, out);
}
